// BaselineMoEModel_71425306132874
// MI455X (gfx1250) — matrix operand order repaired and checked by reading, not compiled, not run on hardware
//
#include <hip/hip_runtime.h>
#include <math.h>

// ---------------- constants ----------------
#define Dm    512
#define NHh   8
#define DHh   64
#define Sq    1024
#define Bb    4
#define NTOK  4096      // B*S
#define NEXPc 8
#define HEXPc 2048
#define NCc   10

typedef __attribute__((ext_vector_type(16))) __bf16 v16bf;
typedef __attribute__((ext_vector_type(8)))  float  v8f;

union FragBF {
    v16bf v;
    uint4 q[2];
};

__device__ __forceinline__ unsigned short f2bf(float f) {
    unsigned int u = __builtin_bit_cast(unsigned int, f);
    unsigned int r = (u + 0x7FFFu + ((u >> 16) & 1u)) >> 16;
    return (unsigned short)r;
}
__device__ __forceinline__ unsigned pack2(float lo, float hi) {
    return (unsigned)f2bf(lo) | ((unsigned)f2bf(hi) << 16);
}

__device__ __forceinline__ v8f wmma_bf16(const FragBF& a, const FragBF& b, v8f c) {
    return __builtin_amdgcn_wmma_f32_16x16x32_bf16(
        false, a.v, false, b.v, (short)0, c, false, false);
}

// ---------------- weight transpose+convert: W[K][N] f32 -> Wt[N][K] bf16 -------
__global__ __launch_bounds__(256) void convert_wt(const float* __restrict__ W,
                                                  unsigned short* __restrict__ Wt,
                                                  int K, int N) {
    int i = blockIdx.x * 256 + threadIdx.x;
    if (i >= K * N) return;
    int k = i / N, n = i - k * N;
    Wt[(size_t)n * K + k] = f2bf(W[i]);
}

// ---------------- embedding + positional encoding ----------------
__global__ __launch_bounds__(256) void embed_kernel(const float* __restrict__ tok,
                                                    float* __restrict__ x) {
    int i = blockIdx.x * 256 + threadIdx.x;
    if (i >= NTOK * Dm) return;
    int c = i & (Dm - 1);
    int n = i >> 9;
    int s = n & (Sq - 1);
    int j = c >> 1;
    float inv = __expf(-(float)(2 * j) * (9.210340371976184f / (float)Dm));
    float ang = (float)s * inv;
    float pe = (c & 1) ? __cosf(ang) : __sinf(ang);
    x[i] = tok[i] * 22.627416997969522f + pe;   // sqrt(512)
}

// ---------------- bf16-WMMA GEMM; B pre-converted bf16 [N][K], loaded from global
// 256 threads = 8 waves; block tile 128x64; each wave 32x32 (4 WMMA accs).
// ACC=1: C[row,:] += rowscale[row*scstride] * (acc + bias)   (fused MoE combine)
template <int RELU, int ACC>
__global__ __launch_bounds__(256) void gemm_tn(
    const float* __restrict__ A, const unsigned short* __restrict__ Bt,
    const float* __restrict__ bias, float* __restrict__ C,
    const float* __restrict__ rowscale,
    int K, int lda, int ldc, int scstride)
{
    __shared__ alignas(16) unsigned int sA[128 * 16];  // 128 rows x 32 bf16

    const int tid  = threadIdx.x;
    const int wave = tid >> 5, lane = tid & 31;
    const int half = lane >> 4, idx = lane & 15;
    const int wm = wave >> 1, wn = wave & 1;
    const int m0 = blockIdx.y * 128, n0 = blockIdx.x * 64;

    v8f acc00 = {}, acc01 = {}, acc10 = {}, acc11 = {};

    const int arow = tid >> 1;            // 0..127
    const int akc  = (tid & 1) * 16;      // 0 or 16

    // per-lane B column base pointers (bf16, row-major over K)
    const uint4* btc0 = (const uint4*)(Bt + (size_t)(n0 + wn * 32 + idx) * K);
    const uint4* btc1 = (const uint4*)(Bt + (size_t)(n0 + wn * 32 + 16 + idx) * K);

    for (int k0 = 0; k0 < K; k0 += 32) {
        { // A tile: 128x32, 16 floats/thread, pack bf16, 2x b128 LDS stores
            const float4* ap = (const float4*)(A + (size_t)(m0 + arow) * lda + k0 + akc);
            float4 a0 = ap[0], a1 = ap[1], a2 = ap[2], a3 = ap[3];
            uint4 p0, p1;
            p0.x = pack2(a0.x, a0.y); p0.y = pack2(a0.z, a0.w);
            p0.z = pack2(a1.x, a1.y); p0.w = pack2(a1.z, a1.w);
            p1.x = pack2(a2.x, a2.y); p1.y = pack2(a2.z, a2.w);
            p1.z = pack2(a3.x, a3.y); p1.w = pack2(a3.z, a3.w);
            uint4* sa4 = (uint4*)sA;
            sa4[arow * 4 + (akc >> 3)]     = p0;
            sa4[arow * 4 + (akc >> 3) + 1] = p1;
        }
        __syncthreads();

        const uint4* sa4 = (const uint4*)sA;
        FragBF af0, af1, bf0, bf1;
        int ar0 = wm * 32 + idx;
        int ar1 = ar0 + 16;
        af0.q[0] = sa4[ar0 * 4 + half * 2];  af0.q[1] = sa4[ar0 * 4 + half * 2 + 1];
        af1.q[0] = sa4[ar1 * 4 + half * 2];  af1.q[1] = sa4[ar1 * 4 + half * 2 + 1];

        int kb = (k0 >> 3) + half * 2;   // uint4 index: k0 + half*16 bf16 elems
        bf0.q[0] = btc0[kb];  bf0.q[1] = btc0[kb + 1];
        bf1.q[0] = btc1[kb];  bf1.q[1] = btc1[kb + 1];

        acc00 = wmma_bf16(af0, bf0, acc00);
        acc01 = wmma_bf16(af0, bf1, acc01);
        acc10 = wmma_bf16(af1, bf0, acc10);
        acc11 = wmma_bf16(af1, bf1, acc11);
        __syncthreads();
    }

    const int colA = n0 + wn * 32 + idx;
    const int colB = colA + 16;
    const float bA = bias[colA];
    const float bB = bias[colB];
#pragma unroll
    for (int r = 0; r < 8; ++r) {
        int row0 = m0 + wm * 32 + half * 8 + r;
        int row1 = row0 + 16;
        if (ACC) {
            float s0 = rowscale[(size_t)row0 * scstride];
            float s1 = rowscale[(size_t)row1 * scstride];
            C[(size_t)row0 * ldc + colA] += s0 * (acc00[r] + bA);
            C[(size_t)row0 * ldc + colB] += s0 * (acc01[r] + bB);
            C[(size_t)row1 * ldc + colA] += s1 * (acc10[r] + bA);
            C[(size_t)row1 * ldc + colB] += s1 * (acc11[r] + bB);
        } else {
            float v00 = acc00[r] + bA, v01 = acc01[r] + bB;
            float v10 = acc10[r] + bA, v11 = acc11[r] + bB;
            if (RELU) {
                v00 = fmaxf(v00, 0.0f); v01 = fmaxf(v01, 0.0f);
                v10 = fmaxf(v10, 0.0f); v11 = fmaxf(v11, 0.0f);
            }
            C[(size_t)row0 * ldc + colA] = v00;
            C[(size_t)row0 * ldc + colB] = v01;
            C[(size_t)row1 * ldc + colA] = v10;
            C[(size_t)row1 * ldc + colB] = v11;
        }
    }
}

// ---------------- flash attention (WMMA, online softmax, WMMA row-sums) --------
__global__ __launch_bounds__(128) void attn_kernel(const float* __restrict__ qkv,
                                                   float* __restrict__ out)
{
    __shared__ alignas(16) unsigned short sK[32 * 64];    // [key][d]
    __shared__ alignas(16) unsigned short sV[64 * 32];    // [d][key] (transposed)
    __shared__ alignas(16) unsigned short sQ[4][16 * 64]; // per-wave Q tile
    __shared__ alignas(16) unsigned short sP[4][16 * 32]; // per-wave P tile

    const int tid  = threadIdx.x;
    const int wave = tid >> 5, lane = tid & 31;
    const int half = lane >> 4, idx = lane & 15;
    const int b = blockIdx.z, head = blockIdx.y, qt = blockIdx.x;
    const int q0 = qt * 64 + wave * 16;

    { // per-wave Q tile (pre-scaled by 1/sqrt(dh))
        int row = lane >> 1;
        int c0  = (lane & 1) * 32;
        const float4* qp = (const float4*)(qkv + (size_t)(b * Sq + q0 + row) * (3 * Dm)
                                           + head * DHh + c0);
        uint4* q4 = (uint4*)sQ[wave];
#pragma unroll
        for (int j = 0; j < 4; ++j) {
            float4 f0 = qp[2 * j], f1 = qp[2 * j + 1];
            uint4 p;
            p.x = pack2(f0.x * 0.125f, f0.y * 0.125f);
            p.y = pack2(f0.z * 0.125f, f0.w * 0.125f);
            p.z = pack2(f1.x * 0.125f, f1.y * 0.125f);
            p.w = pack2(f1.z * 0.125f, f1.w * 0.125f);
            q4[row * 8 + (c0 >> 3) + j] = p;
        }
    }
    __syncthreads();

    FragBF qf0, qf1;
    {
        const uint4* q4 = (const uint4*)sQ[wave];
        qf0.q[0] = q4[idx * 8 + half * 2];
        qf0.q[1] = q4[idx * 8 + half * 2 + 1];
        qf1.q[0] = q4[idx * 8 + 4 + half * 2];
        qf1.q[1] = q4[idx * 8 + 5 + half * 2];
    }

    FragBF onesf;   // all-ones bf16 B fragment: rowsum(P) = P @ ones
    {
        uint4 o;
        o.x = o.y = o.z = o.w = 0x3F803F80u;
        onesf.q[0] = o;
        onesf.q[1] = o;
    }

    v8f o0 = {}, o1 = {}, o2 = {}, o3 = {}, lacc = {};
    float mrow[8];
#pragma unroll
    for (int r = 0; r < 8; ++r) mrow[r] = -1e30f;

    for (int j0 = 0; j0 < Sq; j0 += 32) {
        __syncthreads();
        { // cooperative K (b128) and V (transposed) tile loads
            int key = tid >> 2;
            int c0  = (tid & 3) * 16;
            const float4* kp = (const float4*)(qkv + (size_t)(b * Sq + j0 + key) * (3 * Dm)
                                               + Dm + head * DHh + c0);
            const float4* vp = (const float4*)(qkv + (size_t)(b * Sq + j0 + key) * (3 * Dm)
                                               + 2 * Dm + head * DHh + c0);
            float4 k0v = kp[0], k1v = kp[1], k2v = kp[2], k3v = kp[3];
            uint4 pk0, pk1;
            pk0.x = pack2(k0v.x, k0v.y); pk0.y = pack2(k0v.z, k0v.w);
            pk0.z = pack2(k1v.x, k1v.y); pk0.w = pack2(k1v.z, k1v.w);
            pk1.x = pack2(k2v.x, k2v.y); pk1.y = pack2(k2v.z, k2v.w);
            pk1.z = pack2(k3v.x, k3v.y); pk1.w = pack2(k3v.z, k3v.w);
            uint4* k4s = (uint4*)sK;
            k4s[key * 8 + (c0 >> 3)]     = pk0;
            k4s[key * 8 + (c0 >> 3) + 1] = pk1;

            float4 v0 = vp[0], v1 = vp[1], v2 = vp[2], v3 = vp[3];
            float vv[16] = { v0.x, v0.y, v0.z, v0.w, v1.x, v1.y, v1.z, v1.w,
                             v2.x, v2.y, v2.z, v2.w, v3.x, v3.y, v3.z, v3.w };
#pragma unroll
            for (int j = 0; j < 16; ++j)
                sV[(c0 + j) * 32 + key] = f2bf(vv[j]);
        }
        __syncthreads();

        const uint4* k4 = (const uint4*)sK;
        FragBF kf;
        v8f s0 = {}, s1 = {};
        kf.q[0] = k4[idx * 8 + half * 2];     kf.q[1] = k4[idx * 8 + half * 2 + 1];
        s0 = wmma_bf16(qf0, kf, s0);
        kf.q[0] = k4[idx * 8 + 4 + half * 2]; kf.q[1] = k4[idx * 8 + 5 + half * 2];
        s0 = wmma_bf16(qf1, kf, s0);
        int i2 = idx + 16;
        kf.q[0] = k4[i2 * 8 + half * 2];      kf.q[1] = k4[i2 * 8 + half * 2 + 1];
        s1 = wmma_bf16(qf0, kf, s1);
        kf.q[0] = k4[i2 * 8 + 4 + half * 2];  kf.q[1] = k4[i2 * 8 + 5 + half * 2];
        s1 = wmma_bf16(qf1, kf, s1);

        // online softmax: row max via 16-lane xor-shuffles; row sum via WMMA(P,ones)
#pragma unroll
        for (int r = 0; r < 8; ++r) {
            float c = fmaxf(s0[r], s1[r]);
#pragma unroll
            for (int mm = 1; mm < 16; mm <<= 1)
                c = fmaxf(c, __shfl_xor(c, mm, 32));
            float mn = fmaxf(mrow[r], c);
            float alpha = __expf(mrow[r] - mn);
            mrow[r] = mn;
            float p0 = __expf(s0[r] - mn);
            float p1 = __expf(s1[r] - mn);
            o0[r] *= alpha; o1[r] *= alpha; o2[r] *= alpha; o3[r] *= alpha;
            lacc[r] *= alpha;
            int rr = half * 8 + r;
            sP[wave][rr * 32 + idx]      = f2bf(p0);
            sP[wave][rr * 32 + idx + 16] = f2bf(p1);
        }
        asm volatile("s_wait_dscnt 0x0" ::: "memory");  // per-wave P staging

        const uint4* p4 = (const uint4*)sP[wave];
        FragBF pf;
        pf.q[0] = p4[idx * 4 + half * 2];
        pf.q[1] = p4[idx * 4 + half * 2 + 1];

        lacc = wmma_bf16(pf, onesf, lacc);   // row sums broadcast to all lanes

        const uint4* v4 = (const uint4*)sV;
        FragBF vf;
        {   int d = 0 * 16 + idx;
            vf.q[0] = v4[d * 4 + half * 2]; vf.q[1] = v4[d * 4 + half * 2 + 1];
            o0 = wmma_bf16(pf, vf, o0); }
        {   int d = 1 * 16 + idx;
            vf.q[0] = v4[d * 4 + half * 2]; vf.q[1] = v4[d * 4 + half * 2 + 1];
            o1 = wmma_bf16(pf, vf, o1); }
        {   int d = 2 * 16 + idx;
            vf.q[0] = v4[d * 4 + half * 2]; vf.q[1] = v4[d * 4 + half * 2 + 1];
            o2 = wmma_bf16(pf, vf, o2); }
        {   int d = 3 * 16 + idx;
            vf.q[0] = v4[d * 4 + half * 2]; vf.q[1] = v4[d * 4 + half * 2 + 1];
            o3 = wmma_bf16(pf, vf, o3); }
    }

#pragma unroll
    for (int r = 0; r < 8; ++r) {
        float invl = 1.0f / lacc[r];
        int row = b * Sq + q0 + half * 8 + r;
        float* op = out + (size_t)row * Dm + head * DHh + idx;
        op[0]  = o0[r] * invl;
        op[16] = o1[r] * invl;
        op[32] = o2[r] * invl;
        op[48] = o3[r] * invl;
    }
}

// ---------------- residual + LayerNorm ----------------
__global__ __launch_bounds__(256) void ln_kernel(const float* __restrict__ x,
                                                 const float* __restrict__ add,
                                                 const float* __restrict__ g,
                                                 const float* __restrict__ bta,
                                                 float* __restrict__ out)
{
    __shared__ float red[256];
    const int row = blockIdx.x;
    const int tid = threadIdx.x;
    const size_t base = (size_t)row * Dm;
    float v0 = x[base + tid]       + (add ? add[base + tid]       : 0.0f);
    float v1 = x[base + tid + 256] + (add ? add[base + tid + 256] : 0.0f);
    red[tid] = v0 + v1;
    __syncthreads();
    for (int s = 128; s > 0; s >>= 1) {
        if (tid < s) red[tid] += red[tid + s];
        __syncthreads();
    }
    float mu = red[0] * (1.0f / Dm);
    __syncthreads();
    float d0 = v0 - mu, d1 = v1 - mu;
    red[tid] = d0 * d0 + d1 * d1;
    __syncthreads();
    for (int s = 128; s > 0; s >>= 1) {
        if (tid < s) red[tid] += red[tid + s];
        __syncthreads();
    }
    float rs = rsqrtf(red[0] * (1.0f / Dm) + 1e-5f);
    out[base + tid]       = d0 * rs * g[tid]       + bta[tid];
    out[base + tid + 256] = d1 * rs * g[tid + 256] + bta[tid + 256];
}

// ---------------- router: fused logits + softmax gates + top-2 + combine ------
__global__ __launch_bounds__(256) void router_kernel(const float* __restrict__ x,
                                                     const float* __restrict__ rw,
                                                     const float* __restrict__ rb,
                                                     float* __restrict__ gates_out,
                                                     float* __restrict__ idx_out,
                                                     float* __restrict__ scales)
{
    int n = blockIdx.x * 256 + threadIdx.x;
    if (n >= NTOK) return;
    float lg[NEXPc];
#pragma unroll
    for (int e = 0; e < NEXPc; ++e) lg[e] = rb[e];
    const float4* xp = (const float4*)(x + (size_t)n * Dm);
    for (int k = 0; k < Dm; k += 4) {
        float4 xv = xp[k >> 2];
        const float* w0 = rw + (size_t)k * NEXPc;
#pragma unroll
        for (int e = 0; e < NEXPc; ++e)
            lg[e] += xv.x * w0[e] + xv.y * w0[NEXPc + e]
                   + xv.z * w0[2 * NEXPc + e] + xv.w * w0[3 * NEXPc + e];
    }
    float mx = lg[0];
#pragma unroll
    for (int e = 1; e < NEXPc; ++e) mx = fmaxf(mx, lg[e]);
    float ssum = 0.0f;
#pragma unroll
    for (int e = 0; e < NEXPc; ++e) ssum += __expf(lg[e] - mx);
    float inv = 1.0f / ssum;
#pragma unroll
    for (int e = 0; e < NEXPc; ++e) gates_out[(size_t)n * NEXPc + e] = __expf(lg[e] - mx) * inv;
    int i0 = 0;
#pragma unroll
    for (int e = 1; e < NEXPc; ++e) if (lg[e] > lg[i0]) i0 = e;
    int i1 = (i0 == 0) ? 1 : 0;
#pragma unroll
    for (int e = 0; e < NEXPc; ++e) if (e != i0 && lg[e] > lg[i1]) i1 = e;
    float e1 = __expf(lg[i1] - lg[i0]);
    float p0 = 1.0f / (1.0f + e1);
    float p1 = e1 * p0;
#pragma unroll
    for (int e = 0; e < NEXPc; ++e) scales[(size_t)n * NEXPc + e] = 0.0f;
    scales[(size_t)n * NEXPc + i0] = p0;
    scales[(size_t)n * NEXPc + i1] = p1;
    idx_out[n * 2 + 0] = (float)i0;
    idx_out[n * 2 + 1] = (float)i1;
}

// ---------------- mean pool over sequence ----------------
__global__ __launch_bounds__(256) void pool_kernel(const float* __restrict__ xln,
                                                   float* __restrict__ pooled)
{
    int i = blockIdx.x * 256 + threadIdx.x;   // 4*512
    if (i >= Bb * Dm) return;
    int b = i >> 9, d = i & (Dm - 1);
    float s = 0.0f;
    const float* p = xln + (size_t)b * Sq * Dm + d;
    for (int t = 0; t < Sq; ++t) s += p[(size_t)t * Dm];
    pooled[i] = s * (1.0f / Sq);
}

// ---------------- classifier ----------------
__global__ __launch_bounds__(64) void logits_kernel(const float* __restrict__ pooled,
                                                    const float* __restrict__ ow,
                                                    const float* __restrict__ ob,
                                                    float* __restrict__ out)
{
    int t = threadIdx.x;
    if (t >= Bb * NCc) return;
    int b = t / NCc, c = t % NCc;
    float s = ob[c];
    for (int k = 0; k < Dm; ++k) s += pooled[b * Dm + k] * ow[k * NCc + c];
    out[b * NCc + c] = s;
}

// ---------------- host side ----------------
static void launch_gemm(const float* A, const unsigned short* Bt, const float* bias,
                        float* C, int M, int N, int K, int relu,
                        const float* rowscale, int acc, hipStream_t stream)
{
    dim3 grid(N / 64, M / 128);
    if (acc)
        gemm_tn<0, 1><<<grid, 256, 0, stream>>>(A, Bt, bias, C, rowscale, K, K, N, NEXPc);
    else if (relu)
        gemm_tn<1, 0><<<grid, 256, 0, stream>>>(A, Bt, bias, C, rowscale, K, K, N, NEXPc);
    else
        gemm_tn<0, 0><<<grid, 256, 0, stream>>>(A, Bt, bias, C, rowscale, K, K, N, NEXPc);
}

static void convert_weight(const float* W, unsigned short* Wt, int K, int N,
                           hipStream_t stream)
{
    int total = K * N;
    convert_wt<<<(total + 255) / 256, 256, 0, stream>>>(W, Wt, K, N);
}

extern "C" void kernel_launch(void* const* d_in, const int* in_sizes, int n_in,
                              void* d_out, int out_size, void* d_ws, size_t ws_size,
                              hipStream_t stream) {
    const float* tok      = (const float*)d_in[0];
    const float* m_wqkv   = (const float*)d_in[1];
    const float* m_bqkv   = (const float*)d_in[2];
    const float* m_wo     = (const float*)d_in[3];
    const float* m_bo     = (const float*)d_in[4];
    const float* m_ln1g   = (const float*)d_in[5];
    const float* m_ln1b   = (const float*)d_in[6];
    const float* m_ln2g   = (const float*)d_in[7];
    const float* m_ln2b   = (const float*)d_in[8];
    const float* rw       = (const float*)d_in[9];
    const float* rb       = (const float*)d_in[10];
    const float* mw1      = (const float*)d_in[11];
    const float* mb1      = (const float*)d_in[12];
    const float* mw2      = (const float*)d_in[13];
    const float* mb2      = (const float*)d_in[14];
    const float* t_wqkv   = (const float*)d_in[15];
    const float* t_bqkv   = (const float*)d_in[16];
    const float* t_wo     = (const float*)d_in[17];
    const float* t_bo     = (const float*)d_in[18];
    const float* t_ln1g   = (const float*)d_in[19];
    const float* t_ln1b   = (const float*)d_in[20];
    const float* t_ln2g   = (const float*)d_in[21];
    const float* t_ln2b   = (const float*)d_in[22];
    const float* t_w1     = (const float*)d_in[23];
    const float* t_b1     = (const float*)d_in[24];
    const float* t_w2     = (const float*)d_in[25];
    const float* t_b2     = (const float*)d_in[26];
    const float* fg       = (const float*)d_in[27];
    const float* fbta     = (const float*)d_in[28];
    const float* ow       = (const float*)d_in[29];
    const float* ob       = (const float*)d_in[30];
    (void)in_sizes; (void)n_in; (void)out_size; (void)ws_size;

    // ---- f32 workspace ----
    float* ws        = (float*)d_ws;
    float* x         = ws;                              // [4096,512]
    float* qkv       = x + (size_t)NTOK * Dm;           // [4096,1536]
    float* ao        = qkv + (size_t)NTOK * 3 * Dm;     // [4096,512]
    float* proj      = ao + (size_t)NTOK * Dm;          // [4096,512]
    float* hbuf      = proj + (size_t)NTOK * Dm;        // [4096,2048]
    float* moe_out   = hbuf + (size_t)NTOK * HEXPc;     // [4096,512]
    float* scales    = moe_out + (size_t)NTOK * Dm;     // [4096,8]
    float* pooled    = scales + (size_t)NTOK * NEXPc;   // [4,512]

    // ---- bf16 transposed weight cache ----
    unsigned short* wcur = (unsigned short*)(pooled + Bb * Dm);
    auto take = [&](size_t n) { unsigned short* p = wcur; wcur += n; return p; };

    unsigned short *b_mqkv[2], *b_mwo[2], *b_w1[2][NEXPc], *b_w2[2][NEXPc];
    unsigned short *b_tqkv[2], *b_two[2], *b_tw1[2], *b_tw2[2];
    for (int i = 0; i < 2; ++i) {
        b_mqkv[i] = take((size_t)Dm * 3 * Dm);
        convert_weight(m_wqkv + (size_t)i * Dm * 3 * Dm, b_mqkv[i], Dm, 3 * Dm, stream);
        b_mwo[i] = take((size_t)Dm * Dm);
        convert_weight(m_wo + (size_t)i * Dm * Dm, b_mwo[i], Dm, Dm, stream);
        for (int e = 0; e < NEXPc; ++e) {
            b_w1[i][e] = take((size_t)Dm * HEXPc);
            convert_weight(mw1 + ((size_t)(i * NEXPc + e)) * Dm * HEXPc,
                           b_w1[i][e], Dm, HEXPc, stream);
            b_w2[i][e] = take((size_t)HEXPc * Dm);
            convert_weight(mw2 + ((size_t)(i * NEXPc + e)) * HEXPc * Dm,
                           b_w2[i][e], HEXPc, Dm, stream);
        }
        b_tqkv[i] = take((size_t)Dm * 3 * Dm);
        convert_weight(t_wqkv + (size_t)i * Dm * 3 * Dm, b_tqkv[i], Dm, 3 * Dm, stream);
        b_two[i] = take((size_t)Dm * Dm);
        convert_weight(t_wo + (size_t)i * Dm * Dm, b_two[i], Dm, Dm, stream);
        b_tw1[i] = take((size_t)Dm * HEXPc);
        convert_weight(t_w1 + (size_t)i * Dm * HEXPc, b_tw1[i], Dm, HEXPc, stream);
        b_tw2[i] = take((size_t)HEXPc * Dm);
        convert_weight(t_w2 + (size_t)i * HEXPc * Dm, b_tw2[i], HEXPc, Dm, stream);
    }

    float* out        = (float*)d_out;
    float* out_logits = out;                            // [4,10]
    float* out_gates  = out + Bb * NCc;                 // [2,4096,8]
    float* out_idx    = out_gates + 2 * NTOK * NEXPc;   // [2,4096,2]

    embed_kernel<<<(NTOK * Dm + 255) / 256, 256, 0, stream>>>(tok, x);

    for (int i = 0; i < 2; ++i) {
        // ===== MoE block =====
        launch_gemm(x, b_mqkv[i], m_bqkv + (size_t)i * 3 * Dm,
                    qkv, NTOK, 3 * Dm, Dm, 0, nullptr, 0, stream);
        attn_kernel<<<dim3(Sq / 64, NHh, Bb), 128, 0, stream>>>(qkv, ao);
        launch_gemm(ao, b_mwo[i], m_bo + (size_t)i * Dm,
                    proj, NTOK, Dm, Dm, 0, nullptr, 0, stream);
        ln_kernel<<<NTOK, 256, 0, stream>>>(x, proj, m_ln1g + i * Dm, m_ln1b + i * Dm, x);

        router_kernel<<<(NTOK + 255) / 256, 256, 0, stream>>>(
            x, rw + (size_t)i * Dm * NEXPc, rb + (size_t)i * NEXPc,
            out_gates + (size_t)i * NTOK * NEXPc,
            out_idx + (size_t)i * NTOK * 2, scales);

        hipMemsetAsync(moe_out, 0, (size_t)NTOK * Dm * sizeof(float), stream);
        for (int e = 0; e < NEXPc; ++e) {
            launch_gemm(x, b_w1[i][e], mb1 + ((size_t)(i * NEXPc + e)) * HEXPc,
                        hbuf, NTOK, HEXPc, Dm, 1, nullptr, 0, stream);
            launch_gemm(hbuf, b_w2[i][e], mb2 + ((size_t)(i * NEXPc + e)) * Dm,
                        moe_out, NTOK, Dm, HEXPc, 0, scales + e, 1, stream);
        }
        ln_kernel<<<NTOK, 256, 0, stream>>>(x, moe_out, m_ln2g + i * Dm, m_ln2b + i * Dm, x);

        // ===== Transformer block =====
        launch_gemm(x, b_tqkv[i], t_bqkv + (size_t)i * 3 * Dm,
                    qkv, NTOK, 3 * Dm, Dm, 0, nullptr, 0, stream);
        attn_kernel<<<dim3(Sq / 64, NHh, Bb), 128, 0, stream>>>(qkv, ao);
        launch_gemm(ao, b_two[i], t_bo + (size_t)i * Dm,
                    proj, NTOK, Dm, Dm, 0, nullptr, 0, stream);
        ln_kernel<<<NTOK, 256, 0, stream>>>(x, proj, t_ln1g + i * Dm, t_ln1b + i * Dm, x);

        launch_gemm(x, b_tw1[i], t_b1 + (size_t)i * HEXPc,
                    hbuf, NTOK, HEXPc, Dm, 1, nullptr, 0, stream);
        launch_gemm(hbuf, b_tw2[i], t_b2 + (size_t)i * Dm,
                    proj, NTOK, Dm, HEXPc, 0, nullptr, 0, stream);
        ln_kernel<<<NTOK, 256, 0, stream>>>(x, proj, t_ln2g + i * Dm, t_ln2b + i * Dm, x);
    }

    ln_kernel<<<NTOK, 256, 0, stream>>>(x, nullptr, fg, fbta, ao);
    pool_kernel<<<(Bb * Dm + 255) / 256, 256, 0, stream>>>(ao, pooled);
    logits_kernel<<<1, 64, 0, stream>>>(pooled, ow, ob, out_logits);
}
